// Self_Attn_89515708383990
// MI455X (gfx1250) — matrix operand order repaired and checked by reading, not compiled, not run on hardware
//
#include <hip/hip_runtime.h>
#include <cstdint>

// ---------------------------------------------------------------------------
// CDNA5 (gfx1250) WMMA bf16 GEMM implementing SAGAN-style self-attention.
// v_wmma_f32_16x16x32_bf16, wave32, LDS double-buffered (even/odd unrolled),
// bumped-pointer staging, v_perm-packed f32->bf16 conversion.
// ---------------------------------------------------------------------------

typedef __attribute__((ext_vector_type(16))) __bf16 v16bf;
typedef __attribute__((ext_vector_type(8)))  float  v8f;

#define TILE_M 128
#define TILE_N 128
#define TILE_K 32
#define LDSW   40          // LDS row stride in ushorts (80B: multiple of 16B, bank de-phasing)

// Pack two f32 into two bf16 (round-half-up) with a single v_perm_b32.
__device__ __forceinline__ unsigned int pack2_bf16(float lo, float hi) {
    unsigned int ul = __float_as_uint(lo) + 0x8000u;
    unsigned int uh = __float_as_uint(hi) + 0x8000u;
    return __builtin_amdgcn_perm(uh, ul, 0x07060302u);
}

struct Frag { union { v16bf v; uint4 q[2]; }; };

// C[m,n] = sum_k A(m,k)*B(k,n) (+bias)(relu)(gamma*.+residual)
// A(m,k) = TRANSA ? A[k*lda+m] : A[m*lda+k];  B row-major [K][ldb].
// Grid (M/128, N/128, batch); 256 threads = 8 waves; wave tile 32x64.
// Requires K % 64 == 0 (nk even) -- true for all GEMMs here.
template <int TRANSA>
__global__ __launch_bounds__(256) void wmma_gemm_kernel(
    const float* __restrict__ A, long long sA, int lda,
    const float* __restrict__ Bm, long long sB, int ldb,
    float*       __restrict__ Cm, long long sC, int ldc,
    const float* __restrict__ bias,
    int do_relu,
    const float* __restrict__ residual, long long sR,
    const float* __restrict__ gammaPtr,
    int K)
{
    __shared__ __align__(16) unsigned short As[2][TILE_M * LDSW];
    __shared__ __align__(16) unsigned short Bs[2][TILE_N * LDSW];

    const int tid  = threadIdx.x;
    const int wave = tid >> 5;
    const int lane = tid & 31;
    const int lrow = lane & 15;
    const int lhi  = lane >> 4;
    const int wm   = (wave & 3) * 32;   // wave tile 32(M) x 64(N)
    const int wn   = (wave >> 2) * 64;
    const int gm   = blockIdx.x * TILE_M;
    const int gn   = blockIdx.y * TILE_N;
    const int b    = blockIdx.z;

    // staging map: one 4x4 sub-block per thread per tile (K-major operands),
    // or 4 row-chunks of 4 (M-major A).
    const int st_k0 = (tid >> 5) * 4;   // 0..28
    const int st_n0 = (tid & 31) * 4;   // 0..124

    // ---- precomputed global offsets + bumped base pointers (no muls in loop)
    const float* aP;
    int aOff[4], aLds[4], aBump;
    if (TRANSA) {
        aP    = A + (long long)b * sA + gm + st_n0;
        aBump = TILE_K * lda;
#pragma unroll
        for (int j = 0; j < 4; ++j) {
            aOff[j] = (st_k0 + j) * lda;
            aLds[j] = (st_n0 + j) * LDSW + st_k0;
        }
    } else {
        aP    = A + (long long)b * sA + (long long)gm * lda;
        aBump = TILE_K;
#pragma unroll
        for (int it = 0; it < 4; ++it) {
            int i4 = tid + it * 256;                      // 1024 float4s per tile
            int m  = i4 >> 3;                             // 8 float4s per row
            int k  = (i4 & 7) * 4;
            aOff[it] = m * lda + k;
            aLds[it] = m * LDSW + k;
        }
    }
    const float* bP = Bm + (long long)b * sB + gn + st_n0;
    const int    bBump = TILE_K * ldb;
    int bOff[4], bLds[4];
#pragma unroll
    for (int j = 0; j < 4; ++j) {
        bOff[j] = (st_k0 + j) * ldb;
        bLds[j] = (st_n0 + j) * LDSW + st_k0;
    }

    float4 ar[4], br[4];
    auto load_tile = [&]() {                 // 8x global_load_b128, then bump
#pragma unroll
        for (int j = 0; j < 4; ++j) ar[j] = *(const float4*)(aP + aOff[j]);
#pragma unroll
        for (int j = 0; j < 4; ++j) br[j] = *(const float4*)(bP + bOff[j]);
        aP += aBump;
        bP += bBump;
    };

    auto store_tile = [&](int buf) {         // pack + 8x ds_store_b64
        unsigned short* Ab = &As[buf][0];
        unsigned short* Bb = &Bs[buf][0];
        if (TRANSA) {
#pragma unroll
            for (int j = 0; j < 4; ++j) {    // register 4x4 transpose
                uint2 v;
                v.x = pack2_bf16(((const float*)&ar[0])[j], ((const float*)&ar[1])[j]);
                v.y = pack2_bf16(((const float*)&ar[2])[j], ((const float*)&ar[3])[j]);
                *(uint2*)&Ab[aLds[j]] = v;
            }
        } else {
#pragma unroll
            for (int it = 0; it < 4; ++it) {
                uint2 v;
                v.x = pack2_bf16(ar[it].x, ar[it].y);
                v.y = pack2_bf16(ar[it].z, ar[it].w);
                *(uint2*)&Ab[aLds[it]] = v;
            }
        }
#pragma unroll
        for (int j = 0; j < 4; ++j) {        // register 4x4 transpose
            uint2 v;
            v.x = pack2_bf16(((const float*)&br[0])[j], ((const float*)&br[1])[j]);
            v.y = pack2_bf16(((const float*)&br[2])[j], ((const float*)&br[3])[j]);
            *(uint2*)&Bb[bLds[j]] = v;
        }
    };

    v8f acc[2][4];
#pragma unroll
    for (int i = 0; i < 2; ++i)
#pragma unroll
        for (int j = 0; j < 4; ++j)
            acc[i][j] = (v8f){0.f, 0.f, 0.f, 0.f, 0.f, 0.f, 0.f, 0.f};

    const int k0a = lhi * 8;    // A frag: lanes 0-15 K{0..7,16..23}; 16-31 K{8..15,24..31}
    const int k0b = lhi * 8;    // B frag: lanes 0-15 K{0..7,16..23}; 16-31 K{8..15,24..31}
    const int aFragRow = (wm + lrow) * LDSW + k0a;
    const int bFragRow = (wn + lrow) * LDSW + k0b;

    auto compute = [&](int buf) {            // 12x ds_load_b128 + 8x wmma
        const unsigned short* Ab = &As[buf][0];
        const unsigned short* Bb = &Bs[buf][0];
        Frag af[2], bfr[4];
#pragma unroll
        for (int mi = 0; mi < 2; ++mi) {
            const unsigned short* p = &Ab[aFragRow + mi * (16 * LDSW)];
            af[mi].q[0] = *(const uint4*)(p);
            af[mi].q[1] = *(const uint4*)(p + 16);
        }
#pragma unroll
        for (int ni = 0; ni < 4; ++ni) {
            const unsigned short* p = &Bb[bFragRow + ni * (16 * LDSW)];
            bfr[ni].q[0] = *(const uint4*)(p);
            bfr[ni].q[1] = *(const uint4*)(p + 16);
        }
#pragma unroll
        for (int mi = 0; mi < 2; ++mi)
#pragma unroll
            for (int ni = 0; ni < 4; ++ni)
                acc[mi][ni] = __builtin_amdgcn_wmma_f32_16x16x32_bf16(
                    false, af[mi].v, false, bfr[ni].v,
                    (short)0, acc[mi][ni], false, false);
    };

    // --- double-buffered pipeline, even/odd unrolled: buffer ids are literals
    const int nk2 = K / (2 * TILE_K);        // nk even for all GEMMs here
    load_tile();
    store_tile(0);
    __syncthreads();
    for (int it = 0; it < nk2 - 1; ++it) {
        load_tile();  compute(0);  store_tile(1);  __syncthreads();
        load_tile();  compute(1);  store_tile(0);  __syncthreads();
    }
    load_tile();  compute(0);  store_tile(1);  __syncthreads();
    compute(1);

    // ---- epilogue: bias / relu / gamma*val + residual, store f32 ----
    const float gscale = (gammaPtr != nullptr) ? gammaPtr[0] : 0.0f;
    const float* res = residual ? residual + (long long)b * sR : nullptr;
    float* Cb = Cm + (long long)b * sC;
#pragma unroll
    for (int mi = 0; mi < 2; ++mi) {
#pragma unroll
        for (int ni = 0; ni < 4; ++ni) {
            const int n = gn + wn + ni * 16 + lrow;
#pragma unroll
            for (int r = 0; r < 8; ++r) {
                const int m = gm + wm + mi * 16 + lhi * 8 + r;
                float v = acc[mi][ni][r];
                if (bias)    v += bias[m];
                if (do_relu) v = fmaxf(v, 0.0f);
                if (res)     v = gscale * v + res[m * ldc + n];
                Cb[m * ldc + n] = v;
            }
        }
    }
}

// Softmax over axis n of attn[b][n][m], in place. One thread per column m
// (coalesced across m). 4 independent online (max,sum) accumulators break the
// serial exp/max dependency chain; merged at the end. 2 reads + 1 write total.
__global__ __launch_bounds__(256) void softmax_axis_n_kernel(
    float* __restrict__ attn, int Nn, int Nm)
{
    const int m = blockIdx.x * 256 + threadIdx.x;
    float* p = attn + (long long)blockIdx.y * Nn * Nm + m;
    float mx[4] = {-3.402823466e38f, -3.402823466e38f,
                   -3.402823466e38f, -3.402823466e38f};
    float s[4] = {0.f, 0.f, 0.f, 0.f};
    for (int n = 0; n < Nn; n += 4) {
#pragma unroll
        for (int j = 0; j < 4; ++j) {
            float v  = p[(n + j) * Nm];
            float nm = fmaxf(mx[j], v);
            s[j]  = s[j] * __expf(mx[j] - nm) + __expf(v - nm);
            mx[j] = nm;
        }
    }
    float M = fmaxf(fmaxf(mx[0], mx[1]), fmaxf(mx[2], mx[3]));
    float S = s[0] * __expf(mx[0] - M) + s[1] * __expf(mx[1] - M) +
              s[2] * __expf(mx[2] - M) + s[3] * __expf(mx[3] - M);
    const float inv = 1.0f / S;
    for (int n = 0; n < Nn; n += 4) {
#pragma unroll
        for (int j = 0; j < 4; ++j) {
            int idx = (n + j) * Nm;
            p[idx] = __expf(p[idx] - M) * inv;
        }
    }
}

extern "C" void kernel_launch(void* const* d_in, const int* in_sizes, int n_in,
                              void* d_out, int out_size, void* d_ws, size_t ws_size,
                              hipStream_t stream) {
    (void)in_sizes; (void)n_in; (void)out_size; (void)ws_size;
    const int Bn = 4, C = 256, Ch = 128, N = 4096;

    const float* x     = (const float*)d_in[0];
    const float* wf1   = (const float*)d_in[1];
    const float* bf1   = (const float*)d_in[2];
    const float* wf2   = (const float*)d_in[3];
    const float* bf2   = (const float*)d_in[4];
    const float* wg1   = (const float*)d_in[5];
    const float* bg1   = (const float*)d_in[6];
    const float* wg2   = (const float*)d_in[7];
    const float* bg2   = (const float*)d_in[8];
    const float* wh    = (const float*)d_in[9];
    const float* bh    = (const float*)d_in[10];
    const float* gamma = (const float*)d_in[11];

    float* out  = (float*)d_out;                 // (B, C, N)
    float* attn = out + (size_t)Bn * C * N;      // (B, N, N)

    float* t = (float*)d_ws;                     // Bn*Ch*N  (t1, then reused for t2)
    float* f = t + (size_t)Bn * Ch * N;
    float* g = f + (size_t)Bn * Ch * N;
    float* h = g + (size_t)Bn * Ch * N;          // Bn*C*N

    const dim3 blk(256);
    const long long sX = (long long)C * N, sT = (long long)Ch * N, sAttn = (long long)N * N;

#define GEMM(TA, Ap, sA_, lda_, Bp, sB_, ldb_, Cp, sC_, ldc_, biasp, relu_, resp, sR_, gp, M_, K_) \
    wmma_gemm_kernel<TA><<<dim3((M_) / 128, N / 128, Bn), blk, 0, stream>>>(                       \
        Ap, sA_, lda_, Bp, sB_, ldb_, Cp, sC_, ldc_, biasp, relu_, resp, sR_, gp, K_)

    // t1 = relu(wf1 @ x + bf1)
    GEMM(0, wf1, 0LL, C,  x, sX, N, t, sT, N, bf1, 1, nullptr, 0LL, nullptr, Ch, C);
    // f  = wf2 @ t1 + bf2
    GEMM(0, wf2, 0LL, Ch, t, sT, N, f, sT, N, bf2, 0, nullptr, 0LL, nullptr, Ch, Ch);
    // t2 = relu(wg1 @ x + bg1)
    GEMM(0, wg1, 0LL, C,  x, sX, N, t, sT, N, bg1, 1, nullptr, 0LL, nullptr, Ch, C);
    // g  = wg2 @ t2 + bg2
    GEMM(0, wg2, 0LL, Ch, t, sT, N, g, sT, N, bg2, 0, nullptr, 0LL, nullptr, Ch, Ch);
    // h  = wh @ x + bh
    GEMM(0, wh,  0LL, C,  x, sX, N, h, sX, N, bh, 0, nullptr, 0LL, nullptr, C, C);
    // attn = f^T @ g    (A transposed: f stored Ch x N, lda = N)
    GEMM(1, f, sT, N, g, sT, N, attn, sAttn, N, nullptr, 0, nullptr, 0LL, nullptr, N, Ch);
    // softmax over query axis n (column softmax), in place in d_out
    softmax_axis_n_kernel<<<dim3(N / 256, Bn), blk, 0, stream>>>(attn, N, N);
    // out = gamma * (h @ attn_soft) + x
    GEMM(0, h, sX, N, attn, sAttn, N, out, sX, N, nullptr, 0, x, sX, gamma, C, N);
#undef GEMM
}
